// Decoder_25082609008658
// MI455X (gfx1250) — matrix operand order repaired and checked by reading, not compiled, not run on hardware
//
#include <hip/hip_runtime.h>
#include <cstdint>
#include <cstddef>

// ---------------- types for CDNA5 WMMA ----------------
typedef __bf16 bf16_t;
typedef __attribute__((ext_vector_type(16))) __bf16 v16bf;
typedef __attribute__((ext_vector_type(8)))  __bf16 v8bf;
typedef __attribute__((ext_vector_type(4)))  __bf16 v4bf;
typedef __attribute__((ext_vector_type(8)))  float  v8f;
typedef __attribute__((ext_vector_type(4)))  float  v4f;

#define B_   128
#define T_   25
#define V_   10000
#define E_   512
#define A_   512
#define H_   512
#define ENC_ 2048
#define P_   196
#define XK_  3072          // fused activation row: ctx(2048) | dec(512) | h(512)
#define XOFF_DEC 2048
#define XOFF_H   2560

__device__ __forceinline__ float sigmf(float x) { return 1.0f / (1.0f + __expf(-x)); }

// ---------------- 0. sort batch by descending caption length (stable) --------
__global__ void sort_prep(const int* __restrict__ captions,    // [B,T]
                          const int* __restrict__ lens,        // [B,1]
                          int* __restrict__ sort_ind, int* __restrict__ lens_s,
                          int* __restrict__ caps_s,
                          float* __restrict__ out_caps, float* __restrict__ out_lens,
                          float* __restrict__ out_sort) {
  __shared__ int sl[B_];
  __shared__ int ssi[B_];   // position -> original index
  int i = threadIdx.x;
  sl[i] = lens[i];
  __syncthreads();
  int li = sl[i];
  int rank = 0;
  for (int j = 0; j < B_; ++j) {
    int lj = sl[j];
    if (lj > li || (lj == li && j < i)) rank++;
  }
  sort_ind[rank] = i;
  lens_s[rank]   = li;
  out_sort[rank] = (float)i;
  out_lens[rank] = (float)li;
  ssi[rank] = i;
  __syncthreads();
  int src = ssi[i];
  for (int k = 0; k < T_; ++k) {
    int v = captions[src * T_ + k];
    caps_s[i * T_ + k]   = v;
    out_caps[i * T_ + k] = (float)v;
  }
}

// ---------------- 1. gather sorted encoder + convert to bf16 -----------------
__global__ void gather_enc(const float* __restrict__ src, const int* __restrict__ sort_ind,
                           bf16_t* __restrict__ dst) {
  long idx = (long)blockIdx.x * blockDim.x + threadIdx.x;
  long total4 = (long)B_ * P_ * ENC_ / 4;
  if (idx >= total4) return;
  long idx4 = idx * 4;
  int  b    = (int)(idx4 / ((long)P_ * ENC_));
  long inn  = idx4 - (long)b * P_ * ENC_;
  v4f v = *(const v4f*)(src + (long)sort_ind[b] * P_ * ENC_ + inn);
  v4bf o; o[0] = (bf16_t)v[0]; o[1] = (bf16_t)v[1]; o[2] = (bf16_t)v[2]; o[3] = (bf16_t)v[3];
  *(v4bf*)(dst + idx4) = o;
}

// ---------------- 2. weight conversions --------------------------------------
__global__ void f32_to_bf16_k(const float* __restrict__ s, bf16_t* __restrict__ d, long n) {
  long i = (long)blockIdx.x * blockDim.x + threadIdx.x;
  if (i < n) d[i] = (bf16_t)s[i];
}

__global__ void build_wcat(const float* __restrict__ W_ih,  // [2048,2560]
                           const float* __restrict__ W_hh,  // [2048,512]
                           bf16_t* __restrict__ wcat) {     // [2048,3072]
  long idx = (long)blockIdx.x * blockDim.x + threadIdx.x;
  if (idx >= (long)(4 * H_) * XK_) return;
  int n = (int)(idx / XK_);
  int k = (int)(idx - (long)n * XK_);
  float v = (k < ENC_ + E_) ? W_ih[(long)n * (ENC_ + E_) + k]
                            : W_hh[(long)n * H_ + (k - (ENC_ + E_))];
  wcat[idx] = (bf16_t)v;
}

__global__ void build_biascat(const float* __restrict__ b_ih, const float* __restrict__ b_hh,
                              float* __restrict__ bc) {
  int i = blockIdx.x * blockDim.x + threadIdx.x;
  if (i < 4 * H_) bc[i] = b_ih[i] + b_hh[i];
}

// ---------------- 3. encoder mean --------------------------------------------
__global__ void enc_mean_k(const bf16_t* __restrict__ enc, bf16_t* __restrict__ mean_bf) {
  int idx = blockIdx.x * blockDim.x + threadIdx.x;   // B*ENC = 262144
  if (idx >= B_ * ENC_) return;
  int b = idx >> 11, e = idx & (ENC_ - 1);
  const bf16_t* p = enc + (long)b * P_ * ENC_ + e;
  float s = 0.f;
  for (int pp = 0; pp < P_; ++pp) s += (float)p[(long)pp * ENC_];
  mean_bf[idx] = (bf16_t)(s * (1.0f / (float)P_));
}

// ---------------- 4. generic WMMA GEMM: Out = act(A @ W.T + bias) ------------
// A: bf16 [batches, Mv, K] with row stride lda, batch stride aStride (elements)
// W: bf16 [N, K] row-major (shared across batches)
// Out: f32, row stride ldo, batch stride oStride
// one wave => one 16x64 output tile (1 A-frag, 4 B-frags, 4 v_wmma per K-step)
__global__ void __launch_bounds__(256)
wmma_gemm(const bf16_t* __restrict__ A, long aStride, int lda,
          const bf16_t* __restrict__ W, const float* __restrict__ bias,
          float* __restrict__ Out, long oStride, int ldo,
          int Mv, int N, int K, int nTilesM, int nTilesN64, int batches, int act) {
  int wave  = (blockIdx.x * (int)blockDim.x + (int)threadIdx.x) >> 5;
  int total = batches * nTilesM * nTilesN64;
  if (wave >= total) return;                 // wave-uniform: EXEC stays all-ones for WMMA
  int lane = threadIdx.x & 31;
  int g = lane >> 4, r = lane & 15;
  int per = nTilesM * nTilesN64;
  int b   = wave / per;
  int rem = wave - b * per;
  int mT  = rem / nTilesN64;
  int nT  = rem - mT * nTilesN64;

  int row = mT * 16 + r; if (row >= Mv) row = Mv - 1;      // clamp padded rows
  const bf16_t* Arow = A + (long)b * aStride + (long)row * lda;

  int nBase = nT * 64;
  const bf16_t* Wr[4];
#pragma unroll
  for (int j = 0; j < 4; ++j) {
    int n = nBase + j * 16 + r; if (n >= N) n = N - 1;     // clamp padded cols
    Wr[j] = W + (long)n * K;
  }

  v8f acc[4] = {v8f{}, v8f{}, v8f{}, v8f{}};
  for (int k0 = 0; k0 < K; k0 += 32) {
    // A fragment: 16 contiguous K, elems 0..15 = K[k0+g*16 ..]
    v8bf a0 = *(const v8bf*)(Arow + k0 + g * 16);
    v8bf a1 = *(const v8bf*)(Arow + k0 + g * 16 + 8);
    v16bf av = __builtin_shufflevector(a0, a1, 0,1,2,3,4,5,6,7,8,9,10,11,12,13,14,15);
    int kb = k0 + g * 16;                                  // B fragment: 16 contiguous K
    v16bf b0 = *(const v16bf*)(Wr[0] + kb);
    v16bf b1 = *(const v16bf*)(Wr[1] + kb);
    v16bf b2 = *(const v16bf*)(Wr[2] + kb);
    v16bf b3 = *(const v16bf*)(Wr[3] + kb);
    acc[0] = __builtin_amdgcn_wmma_f32_16x16x32_bf16(false, av, false, b0, (short)0, acc[0], false, false);
    acc[1] = __builtin_amdgcn_wmma_f32_16x16x32_bf16(false, av, false, b1, (short)0, acc[1], false, false);
    acc[2] = __builtin_amdgcn_wmma_f32_16x16x32_bf16(false, av, false, b2, (short)0, acc[2], false, false);
    acc[3] = __builtin_amdgcn_wmma_f32_16x16x32_bf16(false, av, false, b3, (short)0, acc[3], false, false);
  }

  float* Ob = Out + (long)b * oStride;
#pragma unroll
  for (int j = 0; j < 4; ++j) {
    int n = nBase + j * 16 + r;
    if (n >= N) continue;
    float bv = bias ? bias[n] : 0.f;
#pragma unroll
    for (int rr = 0; rr < 8; ++rr) {
      int m = mT * 16 + g * 8 + rr;                        // C/D layout: M = g*8 + vgpr
      if (m >= Mv) continue;
      float v = acc[j][rr] + bv;
      if (act) v = sigmf(v);
      Ob[(long)m * ldo + n] = v;
    }
  }
}

// ---------------- 5. logits GEMM -> masked store straight into d_out ---------
__global__ void __launch_bounds__(256)
wmma_logits(const bf16_t* __restrict__ A, int lda,        // h (bf16) rows, [128,512]
            const bf16_t* __restrict__ W,                  // lin_W bf16 [V,512]
            const float* __restrict__ bias,                // lin_b
            float* __restrict__ out,                       // preds base [B,T,V]
            const int* __restrict__ lens_s, int t,
            int nTilesN64) {
  int wave  = (blockIdx.x * (int)blockDim.x + (int)threadIdx.x) >> 5;
  int total = 8 * nTilesN64;                               // Mtiles(8) x Ntiles
  if (wave >= total) return;
  int lane = threadIdx.x & 31;
  int g = lane >> 4, r = lane & 15;
  int mT = wave / nTilesN64;
  int nT = wave - mT * nTilesN64;

  int row = mT * 16 + r;
  const bf16_t* Arow = A + (long)row * lda;
  int nBase = nT * 64;
  const bf16_t* Wr[4];
#pragma unroll
  for (int j = 0; j < 4; ++j) {
    int n = nBase + j * 16 + r; if (n >= V_) n = V_ - 1;
    Wr[j] = W + (long)n * H_;
  }

  v8f acc[4] = {v8f{}, v8f{}, v8f{}, v8f{}};
  for (int k0 = 0; k0 < H_; k0 += 32) {
    v8bf a0 = *(const v8bf*)(Arow + k0 + g * 16);
    v8bf a1 = *(const v8bf*)(Arow + k0 + g * 16 + 8);
    v16bf av = __builtin_shufflevector(a0, a1, 0,1,2,3,4,5,6,7,8,9,10,11,12,13,14,15);
    int kb = k0 + g * 16;
#pragma unroll
    for (int j = 0; j < 4; ++j) {
      v16bf bb = *(const v16bf*)(Wr[j] + kb);
      acc[j] = __builtin_amdgcn_wmma_f32_16x16x32_bf16(false, av, false, bb, (short)0, acc[j], false, false);
    }
  }

#pragma unroll
  for (int j = 0; j < 4; ++j) {
    int n = nBase + j * 16 + r;
    if (n >= V_) continue;
    float bv = bias[n];
#pragma unroll
    for (int rr = 0; rr < 8; ++rr) {
      int m = mT * 16 + g * 8 + rr;                        // m == sorted batch index
      bool actv = lens_s[m] > t;
      float v = actv ? (acc[j][rr] + bv) : 0.f;
      out[((long)m * T_ + t) * V_ + n] = v;
    }
  }
}

// ---------------- 6. attention score + softmax over P ------------------------
__global__ void score_softmax(const float* __restrict__ att_enc,   // [B,P,A]
                              const float* __restrict__ att_d,     // [B,A] (bias included)
                              const float* __restrict__ finW,      // [A]
                              const float* __restrict__ finb,      // [1]
                              float* __restrict__ wts) {           // [B,P]
  int b = blockIdx.x;
  __shared__ float sAd[A_];
  __shared__ float sSc[P_];
  __shared__ float red[256];
  for (int i = threadIdx.x; i < A_; i += 256) sAd[i] = att_d[b * A_ + i];
  __syncthreads();
  int wave = threadIdx.x >> 5, lane = threadIdx.x & 31;
  for (int p = wave; p < P_; p += 8) {
    const float* ae = att_enc + ((long)b * P_ + p) * A_;
    float s = 0.f;
    for (int a = lane; a < A_; a += 32) {
      float v = ae[a] + sAd[a];
      v = v > 0.f ? v : 0.f;                 // relu
      s += v * finW[a];
    }
    for (int off = 16; off; off >>= 1) s += __shfl_xor(s, off, 32);
    if (lane == 0) sSc[p] = s + finb[0];
  }
  __syncthreads();
  float v = (threadIdx.x < P_) ? sSc[threadIdx.x] : -1e30f;
  red[threadIdx.x] = v; __syncthreads();
  for (int s2 = 128; s2; s2 >>= 1) { if (threadIdx.x < s2) red[threadIdx.x] = fmaxf(red[threadIdx.x], red[threadIdx.x + s2]); __syncthreads(); }
  float mx = red[0]; __syncthreads();
  float e = (threadIdx.x < P_) ? __expf(sSc[threadIdx.x] - mx) : 0.f;
  red[threadIdx.x] = e; __syncthreads();
  for (int s2 = 128; s2; s2 >>= 1) { if (threadIdx.x < s2) red[threadIdx.x] += red[threadIdx.x + s2]; __syncthreads(); }
  float inv = 1.f / red[0];
  if (threadIdx.x < P_) wts[b * P_ + threadIdx.x] = e * inv;
}

// ---------------- 7. ctx = gate * (wts . enc), written bf16 into xcat --------
__global__ void ctx_gate(const bf16_t* __restrict__ enc,     // [B,P,ENC] bf16 (L2 resident)
                         const float* __restrict__ wts,      // [B,P]
                         const float* __restrict__ gate,     // [B,ENC] (sigmoid applied)
                         bf16_t* __restrict__ xcat) {
  int b = blockIdx.x;
  __shared__ float sw[P_];
  for (int i = threadIdx.x; i < P_; i += 256) sw[i] = wts[b * P_ + i];
  __syncthreads();
  int e0 = threadIdx.x * 8;                                // 256 threads * 8 = 2048
  float acc[8] = {0,0,0,0,0,0,0,0};
  const bf16_t* base = enc + (long)b * P_ * ENC_ + e0;
  for (int p = 0; p < P_; ++p) {
    v8bf ev = *(const v8bf*)(base + (long)p * ENC_);
    float w = sw[p];
#pragma unroll
    for (int i = 0; i < 8; ++i) acc[i] += w * (float)ev[i];
  }
  const float* gp = gate + (long)b * ENC_ + e0;
  bf16_t* op = xcat + (long)b * XK_ + e0;
#pragma unroll
  for (int i = 0; i < 8; ++i) op[i] = (bf16_t)(acc[i] * gp[i]);
}

// ---------------- 8. LSTM pointwise update -----------------------------------
__global__ void lstm_update(const float* __restrict__ gates,   // [B,4H]
                            float* __restrict__ h, float* __restrict__ c,
                            bf16_t* __restrict__ xcat,
                            const int* __restrict__ lens_s, int t) {
  int idx = blockIdx.x * blockDim.x + threadIdx.x;             // B*H = 65536
  if (idx >= B_ * H_) return;
  int b = idx >> 9, j = idx & (H_ - 1);
  const float* g = gates + (long)b * 4 * H_;
  float iv = g[j], fv = g[H_ + j], gv = g[2 * H_ + j], ov = g[3 * H_ + j];
  float cn = sigmf(fv) * c[idx] + sigmf(iv) * tanhf(gv);
  float hn = sigmf(ov) * tanhf(cn);
  bool act = lens_s[b] > t;
  float hO = act ? hn : h[idx];
  float cO = act ? cn : c[idx];
  h[idx] = hO; c[idx] = cO;
  xcat[(long)b * XK_ + XOFF_H + j] = (bf16_t)hO;
}

// ---------------- 9. embedding lookup into xcat dec slot ---------------------
__global__ void embed_step(const float* __restrict__ embW, const int* __restrict__ caps_s,
                           int t, bf16_t* __restrict__ xcat) {
  int idx = blockIdx.x * blockDim.x + threadIdx.x;             // B*E = 65536
  if (idx >= B_ * E_) return;
  int b = idx >> 9, e = idx & (E_ - 1);
  int tok = caps_s[b * T_ + (t - 1)];
  xcat[(long)b * XK_ + XOFF_DEC + e] = (bf16_t)embW[(long)tok * E_ + e];
}

// ---------------- 10. h0 -> xcat bf16, preds[:,0,:] --------------------------
__global__ void h_to_xcat(const float* __restrict__ h, bf16_t* __restrict__ xcat) {
  int idx = blockIdx.x * blockDim.x + threadIdx.x;
  if (idx >= B_ * H_) return;
  int b = idx >> 9, j = idx & (H_ - 1);
  xcat[(long)b * XK_ + XOFF_H + j] = (bf16_t)h[idx];
}

__global__ void preds_t0(float* __restrict__ out) {
  long idx = (long)blockIdx.x * blockDim.x + threadIdx.x;      // B*V
  if (idx >= (long)B_ * V_) return;
  int b = (int)(idx / V_);
  int v = (int)(idx - (long)b * V_);
  out[(long)b * T_ * V_ + v] = (v == 0) ? 1.0f : 0.0f;
}

// =============================================================================
extern "C" void kernel_launch(void* const* d_in, const int* in_sizes, int n_in,
                              void* d_out, int out_size, void* d_ws, size_t ws_size,
                              hipStream_t stream) {
  (void)in_sizes; (void)n_in; (void)out_size; (void)ws_size;
  const int*   captions  = (const int*)  d_in[0];
  const float* enc_in    = (const float*)d_in[1];
  const int*   lens      = (const int*)  d_in[2];
  const float* emb_W     = (const float*)d_in[3];
  const float* att_enc_W = (const float*)d_in[4];
  const float* att_enc_b = (const float*)d_in[5];
  const float* att_dec_W = (const float*)d_in[6];
  const float* att_dec_b = (const float*)d_in[7];
  const float* att_fin_W = (const float*)d_in[8];
  const float* att_fin_b = (const float*)d_in[9];
  const float* f_beta_W  = (const float*)d_in[10];
  const float* f_beta_b  = (const float*)d_in[11];
  const float* W_ih      = (const float*)d_in[12];
  // b_ih = d_in[13], W_hh = d_in[14], b_hh = d_in[15]
  const float* b_ih      = (const float*)d_in[13];
  const float* W_hh      = (const float*)d_in[14];
  const float* b_hh      = (const float*)d_in[15];
  const float* lin_W     = (const float*)d_in[16];
  const float* lin_b     = (const float*)d_in[17];
  const float* h_init_W  = (const float*)d_in[18];
  const float* h_init_b  = (const float*)d_in[19];
  const float* c_init_W  = (const float*)d_in[20];
  const float* c_init_b  = (const float*)d_in[21];

  float* outF = (float*)d_out;
  float* out_caps = outF + (long)B_ * T_ * V_;
  float* out_lens = out_caps + B_ * T_;
  float* out_sort = out_lens + B_;

  // ---- workspace partition (aligned 256B) ----
  char* base = (char*)d_ws;
  size_t off = 0;
  auto alloc = [&](size_t bytes) -> char* {
    char* p = base + off;
    off = (off + bytes + 255) & ~(size_t)255;
    return p;
  };
  int*     sort_ind = (int*)    alloc(B_ * 4);
  int*     lens_s   = (int*)    alloc(B_ * 4);
  int*     caps_s   = (int*)    alloc(B_ * T_ * 4);
  bf16_t*  enc_bf   = (bf16_t*) alloc((size_t)B_ * P_ * ENC_ * 2);     // 102.8 MB (L2-resident)
  float*   att_enc  = (float*)  alloc((size_t)B_ * P_ * A_ * 4);       // 51.4 MB
  bf16_t*  watt_enc = (bf16_t*) alloc((size_t)A_ * ENC_ * 2);
  bf16_t*  watt_dec = (bf16_t*) alloc((size_t)A_ * E_ * 2);
  bf16_t*  wfbeta   = (bf16_t*) alloc((size_t)ENC_ * H_ * 2);
  bf16_t*  whinit   = (bf16_t*) alloc((size_t)H_ * ENC_ * 2);
  bf16_t*  wcinit   = (bf16_t*) alloc((size_t)H_ * ENC_ * 2);
  bf16_t*  wlin     = (bf16_t*) alloc((size_t)V_ * H_ * 2);
  bf16_t*  wcat     = (bf16_t*) alloc((size_t)4 * H_ * XK_ * 2);
  float*   biascat  = (float*)  alloc((size_t)4 * H_ * 4);
  bf16_t*  mean_bf  = (bf16_t*) alloc((size_t)B_ * ENC_ * 2);
  float*   hbuf     = (float*)  alloc((size_t)B_ * H_ * 4);
  float*   cbuf     = (float*)  alloc((size_t)B_ * H_ * 4);
  bf16_t*  xcat     = (bf16_t*) alloc((size_t)B_ * XK_ * 2);
  float*   att_d    = (float*)  alloc((size_t)B_ * A_ * 4);
  float*   wts      = (float*)  alloc((size_t)B_ * P_ * 4);
  float*   gate     = (float*)  alloc((size_t)B_ * ENC_ * 4);
  float*   gatesb   = (float*)  alloc((size_t)B_ * 4 * H_ * 4);

  auto cdiv = [](long a, long b) { return (int)((a + b - 1) / b); };

  // ---- one-time setup ----
  sort_prep<<<1, B_, 0, stream>>>(captions, lens, sort_ind, lens_s, caps_s,
                                  out_caps, out_lens, out_sort);
  gather_enc<<<cdiv((long)B_ * P_ * ENC_ / 4, 256), 256, 0, stream>>>(enc_in, sort_ind, enc_bf);

  f32_to_bf16_k<<<cdiv((long)A_ * ENC_, 256), 256, 0, stream>>>(att_enc_W, watt_enc, (long)A_ * ENC_);
  f32_to_bf16_k<<<cdiv((long)A_ * E_,   256), 256, 0, stream>>>(att_dec_W, watt_dec, (long)A_ * E_);
  f32_to_bf16_k<<<cdiv((long)ENC_ * H_, 256), 256, 0, stream>>>(f_beta_W,  wfbeta,   (long)ENC_ * H_);
  f32_to_bf16_k<<<cdiv((long)H_ * ENC_, 256), 256, 0, stream>>>(h_init_W,  whinit,   (long)H_ * ENC_);
  f32_to_bf16_k<<<cdiv((long)H_ * ENC_, 256), 256, 0, stream>>>(c_init_W,  wcinit,   (long)H_ * ENC_);
  f32_to_bf16_k<<<cdiv((long)V_ * H_,   256), 256, 0, stream>>>(lin_W,     wlin,     (long)V_ * H_);
  build_wcat<<<cdiv((long)4 * H_ * XK_, 256), 256, 0, stream>>>(W_ih, W_hh, wcat);
  build_biascat<<<cdiv(4 * H_, 256), 256, 0, stream>>>(b_ih, b_hh, biascat);

  enc_mean_k<<<cdiv((long)B_ * ENC_, 256), 256, 0, stream>>>(enc_bf, mean_bf);

  // h0 = mean @ h_init_W.T + b ; c0 likewise   (M=128,N=512,K=2048)
  wmma_gemm<<<8, 256, 0, stream>>>(mean_bf, 0, ENC_, whinit, h_init_b,
                                   hbuf, 0, H_, B_, H_, ENC_, 8, 8, 1, 0);
  wmma_gemm<<<8, 256, 0, stream>>>(mean_bf, 0, ENC_, wcinit, c_init_b,
                                   cbuf, 0, H_, B_, H_, ENC_, 8, 8, 1, 0);
  h_to_xcat<<<cdiv(B_ * H_, 256), 256, 0, stream>>>(hbuf, xcat);

  // att_enc = enc @ att_enc_W.T + b   (batched: 128 x [196x2048]@[2048x512])
  {
    int tiles = B_ * 13 * 8;                    // Mtiles=ceil(196/16)=13, Ntiles=512/64=8
    wmma_gemm<<<cdiv(tiles, 8), 256, 0, stream>>>(
        enc_bf, (long)P_ * ENC_, ENC_, watt_enc, att_enc_b,
        att_enc, (long)P_ * A_, A_, P_, A_, ENC_, 13, 8, B_, 0);
  }

  preds_t0<<<cdiv((long)B_ * V_, 256), 256, 0, stream>>>(outF);

  // ---- 24 sequential decode steps ----
  const int nTilesV = (V_ + 63) / 64;           // 157
  for (int t = 1; t < T_; ++t) {
    embed_step<<<cdiv(B_ * E_, 256), 256, 0, stream>>>(emb_W, caps_s, t, xcat);

    // att_d = dec @ att_dec_W.T + b  (128x512x512)
    wmma_gemm<<<8, 256, 0, stream>>>(xcat + XOFF_DEC, 0, XK_, watt_dec, att_dec_b,
                                     att_d, 0, A_, B_, A_, E_, 8, 8, 1, 0);

    // gate = sigmoid(h @ f_beta_W.T + b)  (128x2048x512)
    wmma_gemm<<<32, 256, 0, stream>>>(xcat + XOFF_H, 0, XK_, wfbeta, f_beta_b,
                                      gate, 0, ENC_, B_, ENC_, H_, 8, 32, 1, 1);

    score_softmax<<<B_, 256, 0, stream>>>(att_enc, att_d, att_fin_W, att_fin_b, wts);
    ctx_gate<<<B_, 256, 0, stream>>>(enc_bf, wts, gate, xcat);

    // gates = [ctx|dec|h] @ Wcat.T + (b_ih+b_hh)   (128x2048x3072)
    wmma_gemm<<<32, 256, 0, stream>>>(xcat, 0, XK_, wcat, biascat,
                                      gatesb, 0, 4 * H_, B_, 4 * H_, XK_, 8, 32, 1, 0);

    lstm_update<<<cdiv(B_ * H_, 256), 256, 0, stream>>>(gatesb, hbuf, cbuf, xcat, lens_s, t);

    // logits -> preds[:, t, :] (masked), 128x10000x512
    wmma_logits<<<cdiv(8 * nTilesV, 8), 256, 0, stream>>>(
        xcat + XOFF_H, XK_, wlin, lin_b, outF, lens_s, t, nTilesV);
  }
}
